// RNN_46866683134661
// MI455X (gfx1250) — matrix operand order repaired and checked by reading, not compiled, not run on hardware
//
#include <hip/hip_runtime.h>
#include <hip/hip_bf16.h>
#include <math.h>

// ---------------------------------------------------------------------------
// 3-layer tanh RNN, H=128, B=512, T=1024, D_IN=D_OUT=1.
//   * intermediates (T,B,H) row-major f16 in workspace (2 x 134MB ping-pong)
//   * projection GEMMs via v_wmma_f32_16x16x32_f16, weights f16 in LDS,
//     B-fragments hoisted to registers, 2 row-tiles per iteration,
//     global_prefetch for the next grid-stride chunk
//   * recurrence h = tanh(z_t + h@W_hh^T): 32 persistent blocks (16 batch
//     rows), h + f16 W_hh in LDS, z-tile double-buffered through LDS with
//     one coalesced b128 load per thread per step, 4 WMMA per wave per step
// ---------------------------------------------------------------------------

#define Hdim   128
#define Bdim   512
#define Tdim   1024

typedef __attribute__((ext_vector_type(16))) _Float16 v16h;
typedef __attribute__((ext_vector_type(8)))  _Float16 v8h;
typedef __attribute__((ext_vector_type(8)))  float    v8f;

// A-matrix fragment (16x32 f16, M x K), contiguous K order matching B:
//  lanes 0-15 : row M=lane,    halves 0-15 = K 0..15
//  lanes16-31 : row M=lane-16, halves 0-15 = K 16..31
__device__ __forceinline__ v16h load_a16(const _Float16* base, int ld,
                                         int laneLo, int laneHi, int k0) {
    const _Float16* q = base + (size_t)laneLo * ld + k0 + laneHi * 16;
    union { v16h v; v8h h[2]; } u;
    u.h[0] = *(const v8h*)(q);
    u.h[1] = *(const v8h*)(q + 8);
    return u.v;
}

// B-matrix fragment (32x16 f16, K x N) for OUT = A * W^T, i.e. B[k][n]=W[n][k].
// Column n of B == row n of row-major W: lane n0+laneLo reads 16 contiguous
// halves at W[n][k0 + laneHi*16].  (lanes 0-15: K 0..15, lanes 16-31: K 16..31)
__device__ __forceinline__ v16h load_bT16(const _Float16* W, int ld,
                                          int laneLo, int laneHi,
                                          int n0, int k0) {
    return *(const v16h*)(W + (size_t)(n0 + laneLo) * ld + k0 + laneHi * 16);
}

__device__ __forceinline__ v8f wmma_f16(v16h a, v16h b, v8f c) {
    return __builtin_amdgcn_wmma_f32_16x16x32_f16(false, a, false, b,
                                                  (short)0, c, false, false);
}

// ---------------------------------------------------------------------------
// Layer-0 projection: z[t*B+b][h] = x[b][t]*w_ih0[h] + b_ih0[h] + b_hh0[h]
// ---------------------------------------------------------------------------
__global__ void __launch_bounds__(256)
rnn_proj0_kernel(const float* __restrict__ x,      // (B,T)
                 const float* __restrict__ w0,     // (H,1)
                 const float* __restrict__ bih,
                 const float* __restrict__ bhh,
                 _Float16*   __restrict__ z)       // (T,B,H)
{
    size_t idx = (size_t)blockIdx.x * 256 + threadIdx.x;   // T*B*H elements
    int    h   = (int)(idx & (Hdim - 1));
    size_t row = idx >> 7;                                  // t*B + b
    int    t   = (int)(row >> 9);                           // B == 512
    int    b   = (int)(row & (Bdim - 1));
    float v = x[(size_t)b * Tdim + t] * w0[h] + bih[h] + bhh[h];
    z[idx] = (_Float16)v;
}

// ---------------------------------------------------------------------------
// Projection GEMM for layers 1/2: z[r][n] = sum_k y[r][k]*w[n][k] + bias[n]
// rows R = B*T.  Two 16-row tiles per iteration; all A fragments loaded
// before the WMMA chain; next chunk prefetched.
// ---------------------------------------------------------------------------
__global__ void __launch_bounds__(256)
rnn_proj_kernel(const _Float16* __restrict__ yin,  // (R,H) f16
                const float*    __restrict__ w,    // (H,H) f32 row-major
                const float*    __restrict__ bih,
                const float*    __restrict__ bhh,
                _Float16*       __restrict__ zout, // (R,H) f16
                int R)
{
    __shared__ alignas(64) _Float16 sW[Hdim * Hdim];   // 32 KB
    __shared__ float sBias[Hdim];

    int tid = threadIdx.x;
    for (int i = tid; i < Hdim * Hdim; i += 256) sW[i] = (_Float16)w[i];
    for (int i = tid; i < Hdim; i += 256)        sBias[i] = bih[i] + bhh[i];
    __syncthreads();

    int wave = tid >> 5, lane = tid & 31;
    int laneLo = lane & 15, laneHi = lane >> 4;
    int n0 = wave * 16, n = n0 + laneLo, mBase = laneHi * 8;

    v16h bf[4];
#pragma unroll
    for (int kk = 0; kk < 4; ++kk)
        bf[kk] = load_bT16(sW, Hdim, laneLo, laneHi, n0, kk * 32);
    float bias = sBias[n];

    const int step = gridDim.x * 32;                 // 32 rows per iteration
    for (int row0 = blockIdx.x * 32; row0 < R; row0 += step) {
        // prefetch next chunk (one cacheline per lane; speculative)
        int next = row0 + step;
        if (next < R)
            __builtin_prefetch((const char*)(yin + (size_t)next * Hdim)
                                   + lane * 256, 0, 1);

        const _Float16* a0 = yin + (size_t)row0 * Hdim;
        const _Float16* a1 = a0 + 16 * Hdim;
        v16h af0[4], af1[4];
#pragma unroll
        for (int kk = 0; kk < 4; ++kk) af0[kk] = load_a16(a0, Hdim, laneLo, laneHi, kk * 32);
#pragma unroll
        for (int kk = 0; kk < 4; ++kk) af1[kk] = load_a16(a1, Hdim, laneLo, laneHi, kk * 32);

        v8f acc0 = {}, acc1 = {};
#pragma unroll
        for (int kk = 0; kk < 4; ++kk) acc0 = wmma_f16(af0[kk], bf[kk], acc0);
#pragma unroll
        for (int kk = 0; kk < 4; ++kk) acc1 = wmma_f16(af1[kk], bf[kk], acc1);

        _Float16* p0 = zout + (size_t)(row0 + mBase) * Hdim + n;
        _Float16* p1 = p0 + 16 * Hdim;
#pragma unroll
        for (int r = 0; r < 8; ++r) p0[r * Hdim] = (_Float16)(acc0[r] + bias);
#pragma unroll
        for (int r = 0; r < 8; ++r) p1[r * Hdim] = (_Float16)(acc1[r] + bias);
    }
}

// ---------------------------------------------------------------------------
// Recurrence: for each time step  h = tanh(z_t + h @ W_hh^T).
// One block per 16 batch rows (32 blocks), 8 waves, wave w owns columns
// [16w,16w+16).  z tile for step t+1 is loaded (coalesced b128 per thread)
// while step t computes, staged through a double-buffered LDS tile.
// ---------------------------------------------------------------------------
__global__ void __launch_bounds__(256)
rnn_rec_kernel(const _Float16* __restrict__ z,    // (T,B,H) f16
               const float*    __restrict__ w_hh, // (H,H) f32
               _Float16*       __restrict__ y,    // (T,B,H) f16
               int T)
{
    __shared__ alignas(64) _Float16 sW[Hdim * Hdim];   // 32 KB
    __shared__ alignas(64) _Float16 sH[16 * Hdim];     //  4 KB
    __shared__ alignas(64) _Float16 sZ[2][16 * Hdim];  //  8 KB

    int tid = threadIdx.x;
    for (int i = tid; i < Hdim * Hdim; i += 256) sW[i] = (_Float16)w_hh[i];
    for (int i = tid; i < 16 * Hdim; i += 256)   sH[i] = (_Float16)0.f;

    int b0 = blockIdx.x * 16;
    const int tileU4 = (16 * Hdim) / 8;                // uint4 per z tile (256)
    const int strideU4 = (Bdim * Hdim) / 8;            // uint4 per time step

    // preload z tile for t=0, then advance pointer to tile 1
    const uint4* zq = (const uint4*)(z + (size_t)b0 * Hdim) + tid;
    uint4 gz = *zq;
    ((uint4*)&sZ[0][0])[tid] = gz;
    zq += strideU4;
    (void)tileU4;
    __syncthreads();

    int wave = tid >> 5, lane = tid & 31;
    int laneLo = lane & 15, laneHi = lane >> 4;
    int n0 = wave * 16, n = n0 + laneLo, mBase = laneHi * 8;

    v16h bf[4];
#pragma unroll
    for (int kk = 0; kk < 4; ++kk)
        bf[kk] = load_bT16(sW, Hdim, laneLo, laneHi, n0, kk * 32);

    _Float16* yp = y + ((size_t)b0 + mBase) * Hdim + n;   // step-0 base
    const size_t ystride = (size_t)Bdim * Hdim;

    int cur = 0;
    for (int t = 0; t < T; ++t) {
        // issue coalesced load of next step's z tile (hidden behind WMMA)
        bool have = (t + 1 < T);
        uint4 gnext;
        if (have) gnext = *zq;
        zq += strideU4;

        v8f acc = {};
#pragma unroll
        for (int kk = 0; kk < 4; ++kk) {
            v16h a = load_a16(sH, Hdim, laneLo, laneHi, kk * 32);
            acc = wmma_f16(a, bf[kk], acc);
        }
        __syncthreads();   // all waves done reading sH for this step

#pragma unroll
        for (int r = 0; r < 8; ++r) {
            int m = mBase + r;
            float v  = acc[r] + (float)sZ[cur][m * Hdim + n];
            float th = tanhf(v);
            sH[m * Hdim + n] = (_Float16)th;
            yp[r * Hdim]     = (_Float16)th;   // base + imm offset r*256B
        }
        if (have) ((uint4*)&sZ[cur ^ 1][0])[tid] = gnext;
        __syncthreads();   // sH + next z tile ready for next step

        yp += ystride;
        cur ^= 1;
    }
}

// ---------------------------------------------------------------------------
// Output: out[b] = dot(y[T-1][b][:], w_out) + b_out
// ---------------------------------------------------------------------------
__global__ void __launch_bounds__(256)
rnn_out_kernel(const _Float16* __restrict__ y,     // (T,B,H)
               const float*    __restrict__ w_out, // (1,H)
               const float*    __restrict__ b_out, // (1,)
               float*          __restrict__ out,   // (B,1)
               int T)
{
    int b = blockIdx.x * 256 + threadIdx.x;
    if (b >= Bdim) return;
    const _Float16* p = y + ((size_t)(T - 1) * Bdim + b) * Hdim;
    float s = 0.f;
#pragma unroll 8
    for (int h = 0; h < Hdim; ++h) s += (float)p[h] * w_out[h];
    out[b] = s + b_out[0];
}

// ---------------------------------------------------------------------------
extern "C" void kernel_launch(void* const* d_in, const int* in_sizes, int n_in,
                              void* d_out, int out_size, void* d_ws, size_t ws_size,
                              hipStream_t stream) {
    const float* x     = (const float*)d_in[0];
    const float* w_ih0 = (const float*)d_in[1];
    const float* w_hh0 = (const float*)d_in[2];
    const float* b_ih0 = (const float*)d_in[3];
    const float* b_hh0 = (const float*)d_in[4];
    const float* w_ih1 = (const float*)d_in[5];
    const float* w_hh1 = (const float*)d_in[6];
    const float* b_ih1 = (const float*)d_in[7];
    const float* b_hh1 = (const float*)d_in[8];
    const float* w_ih2 = (const float*)d_in[9];
    const float* w_hh2 = (const float*)d_in[10];
    const float* b_ih2 = (const float*)d_in[11];
    const float* b_hh2 = (const float*)d_in[12];
    const float* w_out = (const float*)d_in[13];
    const float* b_out = (const float*)d_in[14];
    float* out = (float*)d_out;

    const size_t nElem = (size_t)Bdim * Tdim * Hdim;      // 67,108,864
    _Float16* bufZ = (_Float16*)d_ws;                     // 134 MB
    _Float16* bufY = bufZ + nElem;                        // 134 MB
    const int R = Bdim * Tdim;

    // Layer 0
    rnn_proj0_kernel<<<(unsigned)(nElem / 256), 256, 0, stream>>>(
        x, w_ih0, b_ih0, b_hh0, bufZ);
    rnn_rec_kernel<<<Bdim / 16, 256, 0, stream>>>(bufZ, w_hh0, bufY, Tdim);
    // Layer 1
    rnn_proj_kernel<<<2048, 256, 0, stream>>>(bufY, w_ih1, b_ih1, b_hh1, bufZ, R);
    rnn_rec_kernel<<<Bdim / 16, 256, 0, stream>>>(bufZ, w_hh1, bufY, Tdim);
    // Layer 2
    rnn_proj_kernel<<<2048, 256, 0, stream>>>(bufY, w_ih2, b_ih2, b_hh2, bufZ, R);
    rnn_rec_kernel<<<Bdim / 16, 256, 0, stream>>>(bufZ, w_hh2, bufY, Tdim);
    // Readout
    rnn_out_kernel<<<(Bdim + 255) / 256, 256, 0, stream>>>(bufY, w_out, b_out, out, Tdim);
}
